// SelfAttention_78391743086783
// MI455X (gfx1250) — matrix operand order repaired and checked by reading, not compiled, not run on hardware
//
#include <hip/hip_runtime.h>
#include <hip/hip_bf16.h>

typedef __attribute__((ext_vector_type(16))) _Float16 v16h;
typedef __attribute__((ext_vector_type(8)))  _Float16 v8h;
typedef __attribute__((ext_vector_type(8)))  float    v8f;
typedef __attribute__((ext_vector_type(4)))  float    v4f;

#define BATCH 8
#define NTOK  4096      // H*W
#define CCH   256       // C
#define CK    32        // C/8

// ---------------- workspace layout (in _Float16 elements) ----------------
// fh : [B][N][32]   row-major      (Q)
// gh : [B][N][32]   row-major      (K)
// hT : [B][256][N]  d-major        (V, transposed for contiguous B-fragments)
// WfT: [32][256]    WfT[d][c] = Wf[c][d]
// WgT: [32][256]
// WhT: [256][256]   WhT[d][c] = Wh[c][d]
static constexpr size_t OFF_F  = 0;
static constexpr size_t OFF_G  = OFF_F  + (size_t)BATCH * NTOK * CK;
static constexpr size_t OFF_HT = OFF_G  + (size_t)BATCH * NTOK * CK;
static constexpr size_t OFF_WF = OFF_HT + (size_t)BATCH * CCH  * NTOK;
static constexpr size_t OFF_WG = OFF_WF + (size_t)CK * CCH;
static constexpr size_t OFF_WH = OFF_WG + (size_t)CK * CCH;

__device__ __forceinline__ unsigned lds_off(const void* p) {
  // LDS aperture: flat addr low 32 bits == LDS byte address (ISA 10.2)
  return (unsigned)(unsigned long long)(uintptr_t)p;
}

// ---------------- kernel 0: weight transpose + f32 -> f16 ----------------
__global__ void __launch_bounds__(256)
wconv_kernel(const float* __restrict__ wf, const float* __restrict__ wg,
             const float* __restrict__ wh, _Float16* __restrict__ ws) {
  int i = blockIdx.x * 256 + threadIdx.x;          // 0 .. 65535
  if (i < CCH * CK) {                              // wf/wg are [C][C/8]
    int c = i / CK, d = i % CK;
    ws[OFF_WF + (size_t)d * CCH + c] = (_Float16)wf[i];
    ws[OFF_WG + (size_t)d * CCH + c] = (_Float16)wg[i];
  }
  if (i < CCH * CCH) {                             // wh is [C][C]
    int c = i / CCH, d = i % CCH;
    ws[OFF_WH + (size_t)d * CCH + c] = (_Float16)wh[i];
  }
}

// ---------------- kernel 1: 1x1-conv projections via WMMA ----------------
// One wave handles a 16-row tile of the flattened [B*N, C] input.
__global__ void __launch_bounds__(256)
proj_kernel(const float* __restrict__ x, _Float16* __restrict__ ws) {
  const int lane = threadIdx.x & 31;
  const int wv   = threadIdx.x >> 5;
  const int tile = blockIdx.x * 8 + wv;            // 0 .. 2047
  const int row0 = tile * 16;                      // global row in [0, 32768)
  const int ln = lane & 15, hi = lane >> 4;

  // ---- A fragments: x tile 16x256 f32 -> 8 f16 fragments (K-chunks of 32)
  // ISA 16-bit A 16x32 layout: lanes 0-15 hold K 0-7 & 16-23; lanes 16-31
  // hold K 8-15 & 24-31 for the same row M = lane%16.
  const float* xr = x + (size_t)(row0 + ln) * CCH;
  v16h a[8];
#pragma unroll
  for (int c = 0; c < 8; ++c) {
    const float* p0 = xr + c * 32 + 8 * hi;
    const float* p1 = p0 + 16;
    v4f r0a = *(const v4f*)p0,       r0b = *(const v4f*)(p0 + 4);
    v4f r1a = *(const v4f*)p1,       r1b = *(const v4f*)(p1 + 4);
    v16h av;
#pragma unroll
    for (int j = 0; j < 4; ++j) {
      av[j]      = (_Float16)r0a[j];
      av[4 + j]  = (_Float16)r0b[j];
      av[8 + j]  = (_Float16)r1a[j];
      av[12 + j] = (_Float16)r1b[j];
    }
    a[c] = av;
  }

  const v8f vzero = {0.f, 0.f, 0.f, 0.f, 0.f, 0.f, 0.f, 0.f};

  // B-fragment: lane holds column d = d0+ln, K values 8*hi..+7 and
  // 16+8*hi..+7 within the chunk -> two b128 loads from transposed weights.
  auto mm16 = [&](const _Float16* __restrict__ wt, int d0) -> v8f {
    const _Float16* wp = wt + (size_t)(d0 + ln) * CCH + 8 * hi;
    v8f acc = vzero;
#pragma unroll
    for (int c = 0; c < 8; ++c) {
      v8h b0 = *(const v8h*)(wp + c * 32);
      v8h b1 = *(const v8h*)(wp + c * 32 + 16);
      v16h bv;
#pragma unroll
      for (int j = 0; j < 8; ++j) { bv[j] = b0[j]; bv[8 + j] = b1[j]; }
      acc = __builtin_amdgcn_wmma_f32_16x16x32_f16(false, a[c], false, bv,
                                                   (short)0, acc, false, false);
    }
    return acc;
  };

  const int b  = row0 / NTOK;
  const int n0 = row0 % NTOK;

  // ---- hproj: 16 column tiles, stored transposed into hT[b][d][m]
  _Float16* hT = ws + OFF_HT + (size_t)b * CCH * NTOK;
#pragma unroll 1
  for (int t = 0; t < 16; ++t) {
    v8f acc = mm16(ws + OFF_WH, t * 16);
    // C/D layout: VGPR i, lane -> row M = i + 8*hi, col N = ln.
    // Transposed store: lane writes 8 contiguous m-values for channel t*16+ln.
    v8h st;
#pragma unroll
    for (int i = 0; i < 8; ++i) st[i] = (_Float16)acc[i];
    *(v8h*)(hT + (size_t)(t * 16 + ln) * NTOK + n0 + 8 * hi) = st;
  }

  // ---- f and g: 2 column tiles each, stored row-major [row][k]
  _Float16* fh = ws + OFF_F;
  _Float16* gh = ws + OFF_G;
#pragma unroll 1
  for (int k0 = 0; k0 < 32; k0 += 16) {
    v8f accf = mm16(ws + OFF_WF, k0);
    v8f accg = mm16(ws + OFF_WG, k0);
#pragma unroll
    for (int i = 0; i < 8; ++i) {
      size_t r = (size_t)(row0 + i + 8 * hi) * CK + k0 + ln;
      fh[r] = (_Float16)accf[i];
      gh[r] = (_Float16)accg[i];
    }
  }
}

// ---------------- kernel 2: fused flash attention + residual -------------
// One block = 8 waves, all on the SAME batch. K/V chunks are staged into
// LDS with gfx1250 async global->LDS DMA (double-buffered, ASYNCcnt), so
// the 16 MB V matrix is read from L2 once per block instead of once per
// wave (8x less L2 traffic). Each wave owns a 16-query-row tile and runs
// online softmax over all 4096 keys.
__global__ void __launch_bounds__(256)
attn_kernel(const float* __restrict__ x, const float* __restrict__ gamma,
            const _Float16* __restrict__ ws, float* __restrict__ out) {
  __shared__ __align__(16) _Float16 vbuf[2][CCH][32];   // 2 x 16 KB V chunk
  __shared__ __align__(16) _Float16 gbuf[2][32][32];    // 2 x  2 KB K chunk
  __shared__ __align__(16) _Float16 pbuf[8][16 * 32];   // per-wave P staging

  const int tid  = threadIdx.x;
  const int lane = tid & 31;
  const int wv   = tid >> 5;
  const int b    = blockIdx.x >> 5;                 // 32 blocks per batch
  const int n0   = ((((blockIdx.x & 31) << 3) + wv) << 4);
  const int ln = lane & 15, hi = lane >> 4;

  const _Float16* fh = ws + OFF_F  + (size_t)b * NTOK * CK;
  const _Float16* gh = ws + OFF_G  + (size_t)b * NTOK * CK;
  const _Float16* hT = ws + OFF_HT + (size_t)b * CCH  * NTOK;

  // ---- async global->LDS staging of one 32-key chunk (5 ops per thread) --
  auto issue_chunk = [&](int buf, int m0) {
    // K chunk: 32 rows x 32 f16 = 2 KB; one async b64 per thread
    {
      int row = tid >> 3, piece = tid & 7;          // 8 B per thread
      unsigned l = lds_off(&gbuf[buf][row][piece * 4]);
      unsigned long long g =
          (unsigned long long)(uintptr_t)(gh + (size_t)(m0 + row) * CK + piece * 4);
      asm volatile("global_load_async_to_lds_b64 %0, %1, off"
                   :: "v"(l), "v"(g) : "memory");
    }
    // V chunk: 256 rows x 32 f16 = 16 KB; thread tid stages channel d = tid
    {
      unsigned l = lds_off(&vbuf[buf][tid][0]);
      unsigned long long g =
          (unsigned long long)(uintptr_t)(hT + (size_t)tid * NTOK + m0);
      asm volatile("global_load_async_to_lds_b128 %0, %1, off"
                   :: "v"(l), "v"(g) : "memory");
      asm volatile("global_load_async_to_lds_b128 %0, %1, off"
                   :: "v"(l + 16u), "v"(g + 16ull) : "memory");
      asm volatile("global_load_async_to_lds_b128 %0, %1, off"
                   :: "v"(l + 32u), "v"(g + 32ull) : "memory");
      asm volatile("global_load_async_to_lds_b128 %0, %1, off"
                   :: "v"(l + 48u), "v"(g + 48ull) : "memory");
    }
  };

  // ---- Q fragment (A operand, K = 32 = C/8, single fragment) ----
  const _Float16* fr = fh + (size_t)(n0 + ln) * CK + 8 * hi;
  v8h q0 = *(const v8h*)fr;
  v8h q1 = *(const v8h*)(fr + 16);
  v16h aq;
#pragma unroll
  for (int j = 0; j < 8; ++j) { aq[j] = q0[j]; aq[8 + j] = q1[j]; }

  const v8f vzero = {0.f, 0.f, 0.f, 0.f, 0.f, 0.f, 0.f, 0.f};
  float mrow[8], lrow[8];
  v8f acc[16];
#pragma unroll
  for (int i = 0; i < 8; ++i) { mrow[i] = -1e30f; lrow[i] = 0.f; }
#pragma unroll
  for (int t = 0; t < 16; ++t) acc[t] = vzero;

  _Float16* pl = &pbuf[wv][0];
  constexpr int NCHUNK = NTOK / 32;                 // 128

  issue_chunk(0, 0);                                // prologue

#pragma unroll 1
  for (int c = 0; c < NCHUNK; ++c) {
    const int buf = c & 1;
    if (c + 1 < NCHUNK) {
      issue_chunk(buf ^ 1, (c + 1) * 32);
      // current chunk's 5 async ops complete (in-order); next 5 stay in flight
      asm volatile("s_wait_asynccnt 0x5" ::: "memory");
    } else {
      asm volatile("s_wait_asynccnt 0x0" ::: "memory");
    }
    __syncthreads();                                // publish LDS chunk

    // ---- S = Q x K^T : two 16x16 tiles (K-dim = 32), B from LDS
    const _Float16* gr0 = &gbuf[buf][ln][8 * hi];
    const _Float16* gr1 = &gbuf[buf][16 + ln][8 * hi];
    v8h ga0 = *(const v8h*)gr0, ga1 = *(const v8h*)(gr0 + 16);
    v8h gb0 = *(const v8h*)gr1, gb1 = *(const v8h*)(gr1 + 16);
    v16h bg0, bg1;
#pragma unroll
    for (int j = 0; j < 8; ++j) {
      bg0[j] = ga0[j]; bg0[8 + j] = ga1[j];
      bg1[j] = gb0[j]; bg1[8 + j] = gb1[j];
    }
    v8f s0 = __builtin_amdgcn_wmma_f32_16x16x32_f16(false, aq, false, bg0,
                                                    (short)0, vzero, false, false);
    v8f s1 = __builtin_amdgcn_wmma_f32_16x16x32_f16(false, aq, false, bg1,
                                                    (short)0, vzero, false, false);

    // ---- online softmax; rows live in (VGPR idx, lane-half): M = i + 8*hi
    float scale[8];
#pragma unroll
    for (int i = 0; i < 8; ++i) {
      float v = fmaxf(s0[i], s1[i]);
      v = fmaxf(v, __shfl_xor(v, 1));
      v = fmaxf(v, __shfl_xor(v, 2));
      v = fmaxf(v, __shfl_xor(v, 4));
      v = fmaxf(v, __shfl_xor(v, 8));
      float nm = fmaxf(mrow[i], v);
      scale[i] = __expf(mrow[i] - nm);
      mrow[i]  = nm;
      float p0 = __expf(s0[i] - nm);
      float p1 = __expf(s1[i] - nm);
      float rs = p0 + p1;
      rs += __shfl_xor(rs, 1);
      rs += __shfl_xor(rs, 2);
      rs += __shfl_xor(rs, 4);
      rs += __shfl_xor(rs, 8);
      lrow[i] = lrow[i] * scale[i] + rs;
      // stage P (C-layout -> row-major 16x32 in LDS) for A-fragment reload
      pl[(i + 8 * hi) * 32 + ln]      = (_Float16)p0;
      pl[(i + 8 * hi) * 32 + 16 + ln] = (_Float16)p1;
    }
    asm volatile("s_wait_dscnt 0x0" ::: "memory");  // cross-lane LDS visibility

    // reload P in A-fragment layout: 2x ds_load_b128 per lane
    const _Float16* pr = pl + ln * 32 + 8 * hi;
    v8h pa = *(const v8h*)pr;
    v8h pb = *(const v8h*)(pr + 16);
    v16h ap;
#pragma unroll
    for (int j = 0; j < 8; ++j) { ap[j] = pa[j]; ap[8 + j] = pb[j]; }

    // ---- O = O*scale + P x V : 16 accumulating WMMAs, V from shared LDS
#pragma unroll
    for (int t = 0; t < 16; ++t) {
#pragma unroll
      for (int i = 0; i < 8; ++i) acc[t][i] *= scale[i];
      const _Float16* vr = &vbuf[buf][t * 16 + ln][8 * hi];
      v8h v0 = *(const v8h*)vr;
      v8h v1 = *(const v8h*)(vr + 16);
      v16h bv;
#pragma unroll
      for (int j = 0; j < 8; ++j) { bv[j] = v0[j]; bv[8 + j] = v1[j]; }
      acc[t] = __builtin_amdgcn_wmma_f32_16x16x32_f16(false, ap, false, bv,
                                                      (short)0, acc[t], false, false);
    }
    __syncthreads();                 // buffer reuse guard for next async issue
  }

  // ---- epilogue: out = gamma * (acc / l) + x
  const float gm = gamma[0];
  const float* xb = x   + ((size_t)b * NTOK + n0) * CCH;
  float*       ob = out + ((size_t)b * NTOK + n0) * CCH;
  float rinv[8];
#pragma unroll
  for (int i = 0; i < 8; ++i) rinv[i] = 1.0f / lrow[i];
#pragma unroll 1
  for (int t = 0; t < 16; ++t) {
#pragma unroll
    for (int i = 0; i < 8; ++i) {
      size_t r = (size_t)(i + 8 * hi) * CCH + t * 16 + ln;
      ob[r] = gm * acc[t][i] * rinv[i] + xb[r];
    }
  }
}

// ------------------------------- launcher --------------------------------
extern "C" void kernel_launch(void* const* d_in, const int* in_sizes, int n_in,
                              void* d_out, int out_size, void* d_ws, size_t ws_size,
                              hipStream_t stream) {
  const float* x     = (const float*)d_in[0];
  const float* wf    = (const float*)d_in[1];
  const float* wg    = (const float*)d_in[2];
  const float* wh    = (const float*)d_in[3];
  const float* gamma = (const float*)d_in[4];
  float*       out   = (float*)d_out;
  _Float16*    ws    = (_Float16*)d_ws;

  // weights -> transposed f16
  wconv_kernel<<<256, 256, 0, stream>>>(wf, wg, wh, ws);
  // projections: 2048 waves (16-row tiles of [B*N, C]), 8 waves per block
  proj_kernel<<<256, 256, 0, stream>>>(x, ws);
  // fused attention: 256 blocks (8 query tiles each, same batch per block)
  attn_kernel<<<256, 256, 0, stream>>>(x, gamma, ws, out);
}
